// BonzSelfAttention_28630251995193
// MI455X (gfx1250) — compile-verified, matrix operand order checked by reading, not run on hardware
//
#include <hip/hip_runtime.h>
#include <hip/hip_bf16.h>

// ---------------------------------------------------------------------------
// BonzSelfAttention for MI455X (gfx1250): bf16 WMMA everywhere, fp32 accum.
// B=8, N=2048, D=768, K=256, H=12, G=4, DH=64
// ---------------------------------------------------------------------------

typedef __attribute__((ext_vector_type(16))) __bf16 v16bf;
typedef __attribute__((ext_vector_type(8)))  float  v8f;
typedef __attribute__((ext_vector_type(4)))  int    v4i;
typedef unsigned short u16;

#define BN   16384   // B*N
#define DD   768
#define NN   2048
#define KK   256
#define HH   12

#if __has_builtin(__builtin_amdgcn_global_load_async_to_lds_b128)
#define HAVE_ASYNC_LDS 1
#else
#define HAVE_ASYNC_LDS 0
#endif

struct alignas(32) Frag32B { uint4 lo, hi; };

// Native bf16 convert (RNE) — lets clang emit v_cvt instead of manual rounding.
__device__ __forceinline__ u16 f2bf(float f) {
  __bf16 h = (__bf16)f;
  return __builtin_bit_cast(u16, h);
}

// A/B operand fragment: two 16-byte chunks (K-chunk and K-chunk+16) per lane.
__device__ __forceinline__ v16bf ldfrag(const u16* p0, const u16* p1) {
  Frag32B f;
  f.lo = *(const uint4*)p0;
  f.hi = *(const uint4*)p1;
  return __builtin_bit_cast(v16bf, f);
}

__device__ __forceinline__ v8f wmma_bf16(v16bf a, v16bf b, v8f c) {
  // (neg_a, A, neg_b, B, c_mod, C, reuse_a, reuse_b)
  return __builtin_amdgcn_wmma_f32_16x16x32_bf16(false, a, false, b, (short)0, c,
                                                 false, false);
}

// ---------------------------------------------------------------------------
// Kernel 1: fp32 -> bf16 conversions (+ project_k transpose)
// ---------------------------------------------------------------------------
__global__ void k_convert(const float* __restrict__ x,  const float* __restrict__ wq,
                          const float* __restrict__ wk, const float* __restrict__ wout,
                          const float* __restrict__ pk,
                          u16* xbf, u16* wqbf, u16* wkbf, u16* woutbf, u16* pkt) {
  long i = (long)blockIdx.x * blockDim.x + threadIdx.x;
  const long NX = (long)BN * DD;             // 12,582,912
  if (i < NX) xbf[i] = f2bf(x[i]);
  if (i < 147456) { wqbf[i] = f2bf(wq[i]); wkbf[i] = f2bf(wk[i]); }
  if (i < 589824) woutbf[i] = f2bf(wout[i]);
  if (i < 524288) {                          // pkt[kq][n] = pk[n][kq]
    int kq = (int)(i >> 11), n = (int)(i & 2047);
    pkt[i] = f2bf(pk[(long)n * KK + kq]);
  }
}

// ---------------------------------------------------------------------------
// Kernel 2: grouped pointwise conv (block-diagonal GEMM), Q and K_raw.
//   y = x_g @ w_g^T  per 192-wide group.  One 16x16 tile per wave, 6 WMMAs.
//   Q stored row-major bf16; K_raw stored TRANSPOSED krt[b][d][n] bf16.
// ---------------------------------------------------------------------------
__global__ void __launch_bounds__(256)
k_gconv(const u16* __restrict__ xbf, const u16* __restrict__ wqbf,
        const u16* __restrict__ wkbf, u16* __restrict__ qbf, u16* __restrict__ krt) {
  const int lane = threadIdx.x & 31, wave = threadIdx.x >> 5;
  const int lm = lane & 15, lhalf = lane >> 4;
  int tile = blockIdx.x * 8 + wave;          // 0..98303
  int qk = tile / 49152;
  int rem = tile - qk * 49152;
  int ct = rem % 48;                         // output col tile (16 channels)
  int rt = rem / 48;                         // row tile (16 rows of B*N)
  int g  = ct / 12;                          // group
  const u16* wbf = (qk == 0) ? wqbf : wkbf;
  const int m0 = rt * 16;
  const u16* arow = xbf + (long)(m0 + lm) * DD + g * 192;
  const u16* brow = wbf + (long)g * 36864 + (long)(ct * 16 - g * 192 + lm) * 192;

  v8f acc = {};
#pragma unroll
  for (int ks = 0; ks < 6; ++ks) {
    int koff = ks * 32 + lhalf * 8;
    v16bf a = ldfrag(arow + koff, arow + koff + 16);
    v16bf b = ldfrag(brow + koff, brow + koff + 16);
    acc = wmma_bf16(a, b, acc);
  }

  if (qk == 0) {
#pragma unroll
    for (int j = 0; j < 8; ++j) {
      int row = m0 + j + lhalf * 8, col = ct * 16 + lm;
      qbf[(long)row * DD + col] = f2bf(acc[j]);
    }
  } else {
#pragma unroll
    for (int j = 0; j < 8; ++j) {
      int row = m0 + j + lhalf * 8;          // global (b,n) row
      int b = row >> 11, n = row & 2047;
      int d = ct * 16 + lm;
      krt[((long)b * DD + d) * NN + n] = f2bf(acc[j]);
    }
  }
}

// ---------------------------------------------------------------------------
// Kernel 3: k_proj[b,kq,d] = sum_n pk[n,kq] * k_raw[b,n,d]
//   A = pkt[kq][n] (row-major), B = krt[b][d][n] (column d per lane), 64 steps.
// ---------------------------------------------------------------------------
__global__ void __launch_bounds__(256)
k_kproj(const u16* __restrict__ pkt, const u16* __restrict__ krt,
        float* __restrict__ kproj) {
  const int lane = threadIdx.x & 31, wave = threadIdx.x >> 5;
  const int lm = lane & 15, lhalf = lane >> 4;
  int id = blockIdx.x * 8 + wave;            // 0..6143
  int b = id / 768;
  int rem = id - b * 768;
  int mt = rem / 48, nt = rem % 48;

  const u16* arow = pkt + (long)(mt * 16 + lm) * NN;
  const u16* brow = krt + ((long)b * DD + nt * 16 + lm) * NN;

  v8f acc = {};
  for (int ks = 0; ks < 64; ++ks) {
    int koff = ks * 32 + lhalf * 8;
    v16bf a = ldfrag(arow + koff, arow + koff + 16);
    v16bf bf = ldfrag(brow + koff, brow + koff + 16);
    acc = wmma_bf16(a, bf, acc);
  }
#pragma unroll
  for (int j = 0; j < 8; ++j) {
    int kq = mt * 16 + j + lhalf * 8, d = nt * 16 + lm;
    kproj[((long)b * KK + kq) * DD + d] = acc[j];
  }
}

// ---------------------------------------------------------------------------
// Kernel 4: gather keys (faithful permute/view trick) + values^T, to bf16.
//   keys[b,h,kq,dq] = kproj[b, (h*64+dq)&255, 3*kq + ((h*64+dq)>>8)]
//   valt[b,h,dq,kq] = kproj[b, kq, h*64+dq]
// ---------------------------------------------------------------------------
__global__ void k_gather(const float* __restrict__ kproj,
                         u16* __restrict__ keys, u16* __restrict__ valt) {
  long i = (long)blockIdx.x * blockDim.x + threadIdx.x;  // < 1,572,864
  int dq = (int)(i & 63);
  int kq = (int)((i >> 6) & 255);
  int bh = (int)(i >> 14);
  int h = bh % HH, b = bh / HH;
  int c = h * 64 + dq;
  int kk2 = c & 255, dd2 = 3 * kq + (c >> 8);
  keys[i] = f2bf(kproj[((long)b * KK + kk2) * DD + dd2]);
  valt[((long)bh * 64 + dq) * KK + kq] = f2bf(kproj[((long)b * KK + kq) * DD + c]);
}

// ---------------------------------------------------------------------------
// Kernel 5: attention.  Block = (b, h, 128-row chunk), 8 waves x 16 rows.
//   LDS: keys 32KB + values^T 32KB + per-wave prob buffer 64KB = 128KB.
//   K/V staged with GLOBAL_LOAD_ASYNC_TO_LDS_B128 (ASYNCcnt-tracked DMA).
// ---------------------------------------------------------------------------
__global__ void __launch_bounds__(256)
k_attn(const u16* __restrict__ qbf, const u16* __restrict__ keys,
       const u16* __restrict__ valt, u16* __restrict__ ctxbf) {
  extern __shared__ __align__(16) unsigned char smem_raw[];
  u16* keys_s = (u16*)smem_raw;              // [256][64]
  u16* valt_s = keys_s + 16384;              // [64][256]
  u16* pbuf   = valt_s + 16384;              // 8 waves * [16][256]

  const int lane = threadIdx.x & 31, wave = threadIdx.x >> 5;
  const int lm = lane & 15, lhalf = lane >> 4;
  int chunk = blockIdx.x & 15;
  int bh = blockIdx.x >> 4;
  int b = bh / HH, h = bh - b * HH;

  // stage K and V^T into LDS (16B per thread per iter)
  {
    const uint4* kg = (const uint4*)(keys + (long)bh * 16384);
    const uint4* vg = (const uint4*)(valt + (long)bh * 16384);
    uint4* ks_ = (uint4*)keys_s;
    uint4* vs_ = (uint4*)valt_s;
#if HAVE_ASYNC_LDS
    typedef __attribute__((address_space(1))) v4i* gp_t;   // global int4*
    typedef __attribute__((address_space(3))) v4i* lp_t;   // LDS int4*
    for (int i = threadIdx.x; i < 2048; i += 256) {
      __builtin_amdgcn_global_load_async_to_lds_b128((gp_t)(kg + i),
                                                     (lp_t)(ks_ + i), 0, 0);
      __builtin_amdgcn_global_load_async_to_lds_b128((gp_t)(vg + i),
                                                     (lp_t)(vs_ + i), 0, 0);
    }
#if __has_builtin(__builtin_amdgcn_s_wait_asynccnt)
    __builtin_amdgcn_s_wait_asynccnt(0);
#else
    asm volatile("s_wait_asynccnt 0x0" ::: "memory");
#endif
#else
    for (int i = threadIdx.x; i < 2048; i += 256) { ks_[i] = kg[i]; vs_[i] = vg[i]; }
#endif
  }
  __syncthreads();

  const int n0 = chunk * 128 + wave * 16;
  const u16* qrow = qbf + ((long)(b * NN + n0 + lm)) * DD + h * 64;
  v16bf aq0 = ldfrag(qrow + lhalf * 8,      qrow + lhalf * 8 + 16);
  v16bf aq1 = ldfrag(qrow + 32 + lhalf * 8, qrow + 32 + lhalf * 8 + 16);

  v8f acc[16];
#pragma unroll
  for (int t = 0; t < 16; ++t) { v8f z = {}; acc[t] = z; }

  // dots = q @ keys^T : 16 column tiles x 2 k-steps
#pragma unroll
  for (int t = 0; t < 16; ++t) {
    const u16* kr = keys_s + (t * 16 + lm) * 64;
    v16bf b0 = ldfrag(kr + lhalf * 8,      kr + lhalf * 8 + 16);
    v16bf b1 = ldfrag(kr + 32 + lhalf * 8, kr + 32 + lhalf * 8 + 16);
    acc[t] = wmma_bf16(aq0, b0, acc[t]);
    acc[t] = wmma_bf16(aq1, b1, acc[t]);
  }

  // softmax over 256 (16 tiles x 16 lanes per half-wave), scale 1/sqrt(64)
  u16* pw = pbuf + wave * 4096;
#pragma unroll
  for (int j = 0; j < 8; ++j) {
    float m = -3.4e38f;
#pragma unroll
    for (int t = 0; t < 16; ++t) m = fmaxf(m, acc[t][j] * 0.125f);
    m = fmaxf(m, __shfl_xor(m, 1, 32));
    m = fmaxf(m, __shfl_xor(m, 2, 32));
    m = fmaxf(m, __shfl_xor(m, 4, 32));
    m = fmaxf(m, __shfl_xor(m, 8, 32));
    float s = 0.f;
#pragma unroll
    for (int t = 0; t < 16; ++t) {
      float p = __expf(acc[t][j] * 0.125f - m);
      acc[t][j] = p;
      s += p;
    }
    s += __shfl_xor(s, 1, 32);
    s += __shfl_xor(s, 2, 32);
    s += __shfl_xor(s, 4, 32);
    s += __shfl_xor(s, 8, 32);
    float inv = 1.f / s;
    int row = j + lhalf * 8;
#pragma unroll
    for (int t = 0; t < 16; ++t)
      pw[row * 256 + t * 16 + lm] = f2bf(acc[t][j] * inv);
  }

  // ctx = attn @ values : 4 dh tiles x 8 k-steps
  v8f o[4];
#pragma unroll
  for (int t = 0; t < 4; ++t) { v8f z = {}; o[t] = z; }
#pragma unroll
  for (int ks = 0; ks < 8; ++ks) {
    const u16* pr = pw + lm * 256 + ks * 32 + lhalf * 8;
    v16bf ap = ldfrag(pr, pr + 16);
#pragma unroll
    for (int dt = 0; dt < 4; ++dt) {
      const u16* vr = valt_s + (dt * 16 + lm) * 256 + ks * 32 + lhalf * 8;
      v16bf bv = ldfrag(vr, vr + 16);
      o[dt] = wmma_bf16(ap, bv, o[dt]);
    }
  }
#pragma unroll
  for (int dt = 0; dt < 4; ++dt)
#pragma unroll
    for (int j = 0; j < 8; ++j) {
      int row = b * NN + n0 + j + lhalf * 8;
      ctxbf[(long)row * DD + h * 64 + dt * 16 + lm] = f2bf(o[dt][j]);
    }
}

// ---------------------------------------------------------------------------
// Kernel 6: out = ctx @ w_out^T + b_out; y = x + out; LayerNorm(y).
//   Block = 16 rows; 8 waves x 6 col tiles; 24 k-steps. LN via LDS + shfl.
// ---------------------------------------------------------------------------
__global__ void __launch_bounds__(256)
k_out(const u16* __restrict__ ctxbf, const u16* __restrict__ woutbf,
      const float* __restrict__ x, const float* __restrict__ bout,
      const float* __restrict__ gamma, const float* __restrict__ beta,
      float* __restrict__ out) {
  __shared__ float ybuf[16][772];            // padded vs 64-bank LDS
  const int lane = threadIdx.x & 31, wave = threadIdx.x >> 5;
  const int lm = lane & 15, lhalf = lane >> 4;
  const int r0 = blockIdx.x * 16;

  const u16* arow = ctxbf + (long)(r0 + lm) * DD;
  v8f acc[6];
#pragma unroll
  for (int c = 0; c < 6; ++c) { v8f z = {}; acc[c] = z; }

  for (int ks = 0; ks < 24; ++ks) {
    int koff = ks * 32 + lhalf * 8;
    v16bf a = ldfrag(arow + koff, arow + koff + 16);
    __builtin_prefetch(arow + koff + 64, 0, 1);
#pragma unroll
    for (int c = 0; c < 6; ++c) {
      int ct = wave * 6 + c;
      const u16* brow = woutbf + (long)(ct * 16 + lm) * DD + koff;
      v16bf bf = ldfrag(brow, brow + 16);
      acc[c] = wmma_bf16(a, bf, acc[c]);
    }
  }

#pragma unroll
  for (int c = 0; c < 6; ++c) {
    int ct = wave * 6 + c;
#pragma unroll
    for (int j = 0; j < 8; ++j) {
      int row = j + lhalf * 8, col = ct * 16 + lm;
      ybuf[row][col] = acc[c][j] + bout[col] + x[(long)(r0 + row) * DD + col];
    }
  }
  __syncthreads();

  int rr = threadIdx.x >> 4;                 // row 0..15, 16 lanes per row
  int s  = threadIdx.x & 15;
  float sum = 0.f, sq = 0.f;
  for (int c = s; c < DD; c += 16) { float v = ybuf[rr][c]; sum += v; sq += v * v; }
  sum += __shfl_xor(sum, 1, 32); sq += __shfl_xor(sq, 1, 32);
  sum += __shfl_xor(sum, 2, 32); sq += __shfl_xor(sq, 2, 32);
  sum += __shfl_xor(sum, 4, 32); sq += __shfl_xor(sq, 4, 32);
  sum += __shfl_xor(sum, 8, 32); sq += __shfl_xor(sq, 8, 32);
  float mu = sum * (1.f / DD);
  float var = sq * (1.f / DD) - mu * mu;
  float rstd = rsqrtf(var + 1e-12f);
  for (int c = s; c < DD; c += 16)
    out[(long)(r0 + rr) * DD + c] = (ybuf[rr][c] - mu) * rstd * gamma[c] + beta[c];
}

// ---------------------------------------------------------------------------
// Host launcher
// ---------------------------------------------------------------------------
extern "C" void kernel_launch(void* const* d_in, const int* in_sizes, int n_in,
                              void* d_out, int out_size, void* d_ws, size_t ws_size,
                              hipStream_t stream) {
  const float* x     = (const float*)d_in[0];
  const float* wq    = (const float*)d_in[1];
  const float* wk    = (const float*)d_in[2];
  const float* pk    = (const float*)d_in[3];
  const float* wout  = (const float*)d_in[4];
  const float* bout  = (const float*)d_in[5];
  const float* gamma = (const float*)d_in[6];
  const float* beta  = (const float*)d_in[7];
  float* out = (float*)d_out;
  char* ws = (char*)d_ws;

  // 256B-aligned workspace layout (bytes)
  u16*   xbf   = (u16*)(ws + 0);             // 25,165,824
  u16*   wqbf  = (u16*)(ws + 25165824);      //    294,912
  u16*   wkbf  = (u16*)(ws + 25460736);      //    294,912
  u16*   woutb = (u16*)(ws + 25755648);      //  1,179,648
  u16*   pkt   = (u16*)(ws + 26935296);      //  1,048,576
  u16*   qbf   = (u16*)(ws + 27983872);      // 25,165,824
  u16*   krt   = (u16*)(ws + 53149696);      // 25,165,824
  float* kproj = (float*)(ws + 78315520);    //  6,291,456
  u16*   keys  = (u16*)(ws + 84606976);      //  3,145,728
  u16*   valt  = (u16*)(ws + 87752704);      //  3,145,728
  u16*   ctxbf = (u16*)(ws + 90898432);      // 25,165,824  (end 116,064,256)

  (void)hipFuncSetAttribute(reinterpret_cast<const void*>(k_attn),
                            hipFuncAttributeMaxDynamicSharedMemorySize, 131072);

  k_convert<<<49152, 256, 0, stream>>>(x, wq, wk, wout, pk, xbf, wqbf, wkbf, woutb, pkt);
  k_gconv  <<<12288, 256, 0, stream>>>(xbf, wqbf, wkbf, qbf, krt);
  k_kproj  <<<  768, 256, 0, stream>>>(pkt, krt, kproj);
  k_gather <<< 6144, 256, 0, stream>>>(kproj, keys, valt);
  k_attn   <<< 1536, 256, 131072, stream>>>(qbf, keys, valt, ctxbf);
  k_out    <<< 1024, 256, 0, stream>>>(ctxbf, woutb, x, bout, gamma, beta, out);
}
